// Attention_multi_doublevalue_55439437857013
// MI455X (gfx1250) — matrix operand order repaired and checked by reading, not compiled, not run on hardware
//
#include <hip/hip_runtime.h>

typedef __attribute__((ext_vector_type(16))) __bf16 v16bf;
typedef __attribute__((ext_vector_type(8)))  __bf16 v8bf;
typedef __attribute__((ext_vector_type(8)))  float  v8f;

#define DEV static __device__ __forceinline__

constexpr int B_  = 16;
constexpr int N_  = 906;
constexpr int C_  = 768;
constexpr int H_  = 12;
constexpr int D_  = 64;
constexpr int M1_ = 513;
constexpr int NP_ = 928;             // padded key length: 29 blocks of 32
constexpr int MT2_ = (B_ * N_) / 32; // 453 32-row tiles of flattened (B*N)

// ---------------------------------------------------------------------------
// WMMA fragment loaders (wave32).
// A (16xK, K=32 bf16): lane L holds row L%16; halves: K[h*8..h*8+7], K[16+h*8..]
// B (Kx16, K=32 bf16): lane L holds col L%16; halves: K[h*8..h*8+7], K[16+h*8..]
// ---------------------------------------------------------------------------
DEV v16bf load_a(const __bf16* p, int ld) {
  int lane = threadIdx.x & 31;
  const __bf16* q = p + (size_t)(lane & 15) * ld + (lane >> 4) * 8;
  v16bf f;
  *(v8bf*)&f       = *(const v8bf*)q;
  *((v8bf*)&f + 1) = *(const v8bf*)(q + 16);
  return f;
}
DEV v16bf load_b(const __bf16* p, int ld) {
  int lane = threadIdx.x & 31;
  const __bf16* q = p + (size_t)(lane & 15) * ld + (lane >> 4) * 8;
  v16bf f;
  *(v8bf*)&f       = *(const v8bf*)q;
  *((v8bf*)&f + 1) = *(const v8bf*)(q + 16);
  return f;
}
DEV v8f wmma_bf16(v16bf a, v16bf b, v8f c) {
  return __builtin_amdgcn_wmma_f32_16x16x32_bf16(false, a, false, b, (short)0, c,
                                                 false, false);
}

// ---------------------------------------------------------------------------
// Kernel 1: cast x -> bf16, weights -> bf16 transposed (Wt[w][co][ci]).
// ---------------------------------------------------------------------------
__global__ void cvt_kernel(const float* __restrict__ x,
                           const float* __restrict__ Wq, const float* __restrict__ Wk,
                           const float* __restrict__ Wv, const float* __restrict__ Wvc,
                           const float* __restrict__ Wp,
                           __bf16* __restrict__ xb, __bf16* __restrict__ wt) {
  const long NX = (long)B_ * N_ * C_;
  const long NW = (long)C_ * C_;
  const long total = NX + 5 * NW;
  for (long i = (long)blockIdx.x * blockDim.x + threadIdx.x; i < total;
       i += (long)gridDim.x * blockDim.x) {
    if (i < NX) {
      xb[i] = (__bf16)x[i];
    } else {
      long j = i - NX;
      int w = (int)(j / NW);
      int rem = (int)(j - (long)w * NW);
      int co = rem / C_, ci = rem - co * C_;
      const float* W = (w == 0) ? Wq : (w == 1) ? Wk : (w == 2) ? Wv
                      : (w == 3) ? Wvc : Wp;
      wt[(long)w * NW + (long)co * C_ + ci] = (__bf16)W[(long)ci * C_ + co];
    }
  }
}

// ---------------------------------------------------------------------------
// Kernel 2: projections. One wave computes a 32x64 tile of (x @ W[w]) so each
// weight B-fragment load feeds 2 WMMAs.
//   w=0 -> q[b][h][n][d]  (n padded to NP_)
//   w=1 -> k[b][h][n][d]
//   w=2 -> v  scattered into Va (n<M1) / Vv (n>=M1), stored TRANSPOSED [d][n]
//   w=3 -> vc scattered into Vv (n<M1) / Va (n>=M1), stored TRANSPOSED [d][n]
// ---------------------------------------------------------------------------
__global__ __launch_bounds__(32) void proj_kernel(
    const __bf16* __restrict__ xb, const __bf16* __restrict__ wt,
    __bf16* __restrict__ qh, __bf16* __restrict__ kh,
    __bf16* __restrict__ va, __bf16* __restrict__ vv) {
  const int mt = blockIdx.x;      // 0..452 : 32-row tile of flattened (B*N)
  const int h  = blockIdx.y;      // 0..11  : head (64 output columns)
  const int w  = blockIdx.z;      // 0..3   : which weight
  const __bf16* wp = wt + (size_t)w * C_ * C_;

  v8f acc[2][4];
#pragma unroll
  for (int u = 0; u < 2; ++u)
#pragma unroll
    for (int j = 0; j < 4; ++j) acc[u][j] = (v8f){};

  const __bf16* arow = xb + (size_t)mt * 32 * C_;
  const __bf16* brow = wp + (size_t)h * 64 * C_;
  for (int kk = 0; kk < C_ / 32; ++kk) {
    v16bf a0 = load_a(arow + kk * 32, C_);
    v16bf a1 = load_a(arow + (size_t)16 * C_ + kk * 32, C_);
#pragma unroll
    for (int j = 0; j < 4; ++j) {
      v16bf bf = load_b(brow + (size_t)j * 16 * C_ + kk * 32, C_);
      acc[0][j] = wmma_bf16(a0, bf, acc[0][j]);
      acc[1][j] = wmma_bf16(a1, bf, acc[1][j]);
    }
  }

  const int lane = threadIdx.x & 31, half = lane >> 4, col = lane & 15;
#pragma unroll
  for (int u = 0; u < 2; ++u) {
#pragma unroll
    for (int j = 0; j < 4; ++j) {
#pragma unroll
      for (int i = 0; i < 8; ++i) {
        int r  = u * 16 + i + 8 * half;
        int gn = mt * 32 + r;
        int b  = gn / N_;
        int n  = gn - b * N_;
        int cj = j * 16 + col;               // 0..63 within the head
        __bf16 val = (__bf16)acc[u][j][i];
        size_t bh = (size_t)b * H_ + h;
        if (w == 0) {
          qh[(bh * NP_ + n) * D_ + cj] = val;
        } else if (w == 1) {
          kh[(bh * NP_ + n) * D_ + cj] = val;
        } else {
          __bf16* dst = (w == 2) ? ((n < M1_) ? va : vv)
                                 : ((n < M1_) ? vv : va);
          dst[(bh * D_ + cj) * NP_ + n] = val; // transposed [d][n]
        }
      }
    }
  }
}

// ---------------------------------------------------------------------------
// Kernel 3: flash attention. One wave = 32 query rows of one (b,h); K/V
// fragments loaded once per key block feed both 16-row query halves.
// 30 query tasks: t<17 -> modality a (Va, store n<M1), else modality v
// (Vv, store M1<=n<N). 32-row block 16 straddles M1 and is done by both.
// ---------------------------------------------------------------------------
__global__ __launch_bounds__(32) void attn_kernel(
    const __bf16* __restrict__ qh, const __bf16* __restrict__ kh,
    const __bf16* __restrict__ va, const __bf16* __restrict__ vv,
    __bf16* __restrict__ ao) {
  __shared__ __bf16 P[32 * 32];           // probability tile staging (transpose)

  const int t  = blockIdx.x;
  const int h  = blockIdx.y;
  const int b  = blockIdx.z;
  const int md = (t < 17) ? 0 : 1;
  const int qb = (t < 17) ? t : t - 1;    // 32-query block index (block 16 twice)
  const int lane = threadIdx.x & 31, half = lane >> 4, col = lane & 15;

  const size_t bh = (size_t)b * H_ + h;
  const __bf16* qp = qh + (bh * NP_ + (size_t)qb * 32) * D_;
  const __bf16* kp = kh + bh * NP_ * D_;
  const __bf16* vt = ((md == 0) ? va : vv) + bh * D_ * NP_;

  v16bf qa[2][2];
#pragma unroll
  for (int u = 0; u < 2; ++u) {
    qa[u][0] = load_a(qp + (size_t)u * 16 * D_, D_);       // d 0..31
    qa[u][1] = load_a(qp + (size_t)u * 16 * D_ + 32, D_);  // d 32..63
  }

  v8f O[2][4];
  float m[2][8], l[2][8];
#pragma unroll
  for (int u = 0; u < 2; ++u)
#pragma unroll
    for (int i = 0; i < 8; ++i) { m[u][i] = -3.0e38f; l[u][i] = 0.0f; }
#pragma unroll
  for (int u = 0; u < 2; ++u)
#pragma unroll
    for (int j = 0; j < 4; ++j) O[u][j] = (v8f){};

  const float SC  = 0.125f;               // d^-0.5
  const float L2E = 1.44269504f;

  for (int kb = 0; kb < NP_ / 32; ++kb) {
    const int k0 = kb * 32;

    // prefetch next key block of K and V^T (global_prefetch_b8)
    if (kb + 1 < NP_ / 32) {
      __builtin_prefetch(kp + (size_t)(k0 + 32 + lane) * D_, 0, 3);
      __builtin_prefetch(vt + (size_t)lane * NP_ + k0 + 32, 0, 3);
      __builtin_prefetch(vt + (size_t)(lane + 32) * NP_ + k0 + 32, 0, 3);
    }

    // ---- K fragments: shared by both query halves ----
    v16bf kb00 = load_b(kp + (size_t)k0 * D_, D_);
    v16bf kb01 = load_b(kp + (size_t)k0 * D_ + 32, D_);
    v16bf kb10 = load_b(kp + (size_t)(k0 + 16) * D_, D_);
    v16bf kb11 = load_b(kp + (size_t)(k0 + 16) * D_ + 32, D_);

    v8f S[2][2];
#pragma unroll
    for (int u = 0; u < 2; ++u) {
      v8f s0 = {}, s1 = {};
      s0 = wmma_bf16(qa[u][0], kb00, s0);
      s0 = wmma_bf16(qa[u][1], kb01, s0);
      s1 = wmma_bf16(qa[u][0], kb10, s1);
      s1 = wmma_bf16(qa[u][1], kb11, s1);
      S[u][0] = s0; S[u][1] = s1;
    }

    const bool oob0 = (k0 + col)      >= N_;
    const bool oob1 = (k0 + 16 + col) >= N_;

    // ---- online softmax per query half ----
#pragma unroll
    for (int u = 0; u < 2; ++u) {
      float t0[8], t1[8], mx[8];
#pragma unroll
      for (int i = 0; i < 8; ++i) {
        t0[i] = oob0 ? -3.0e38f : S[u][0][i] * SC;
        t1[i] = oob1 ? -3.0e38f : S[u][1][i] * SC;
        mx[i] = fmaxf(t0[i], t1[i]);
      }
#pragma unroll
      for (int d_ = 1; d_ < 16; d_ <<= 1)
#pragma unroll
        for (int i = 0; i < 8; ++i)
          mx[i] = fmaxf(mx[i], __shfl_xor(mx[i], d_, 32));

      float corr[8], rs[8];
#pragma unroll
      for (int i = 0; i < 8; ++i) {
        float mn = fmaxf(m[u][i], mx[i]);
        corr[i]  = exp2f((m[u][i] - mn) * L2E);
        m[u][i]  = mn;
        t0[i]    = exp2f((t0[i] - mn) * L2E);
        t1[i]    = exp2f((t1[i] - mn) * L2E);
        rs[i]    = t0[i] + t1[i];
      }
#pragma unroll
      for (int d_ = 1; d_ < 16; d_ <<= 1)
#pragma unroll
        for (int i = 0; i < 8; ++i)
          rs[i] += __shfl_xor(rs[i], d_, 32);
#pragma unroll
      for (int i = 0; i < 8; ++i) {
        l[u][i] = l[u][i] * corr[i] + rs[i];
        O[u][0][i] *= corr[i]; O[u][1][i] *= corr[i];
        O[u][2][i] *= corr[i]; O[u][3][i] *= corr[i];
      }

      // D-layout -> A-layout transpose through LDS
#pragma unroll
      for (int i = 0; i < 8; ++i) {
        int r = u * 16 + i + 8 * half;
        P[r * 32 + col]      = (__bf16)t0[i];
        P[r * 32 + 16 + col] = (__bf16)t1[i];
      }
    }

    v16bf pa0 = load_a(P, 32);
    v16bf pa1 = load_a(P + 16 * 32, 32);

    // ---- O += P @ V^T-tiles; V fragments shared by both halves ----
#pragma unroll
    for (int j = 0; j < 4; ++j) {
      v16bf vb = load_b(vt + (size_t)j * 16 * NP_ + k0, NP_);
      O[0][j] = wmma_bf16(pa0, vb, O[0][j]);
      O[1][j] = wmma_bf16(pa1, vb, O[1][j]);
    }
  }

  // normalize and store rows belonging to this task's modality
#pragma unroll
  for (int u = 0; u < 2; ++u) {
#pragma unroll
    for (int i = 0; i < 8; ++i) {
      float inv = 1.0f / l[u][i];
      int r = u * 16 + i + 8 * half;
      int n = qb * 32 + r;
      bool ok = (n < N_) && ((md == 0) ? (n < M1_) : (n >= M1_));
      if (ok) {
        size_t base = ((size_t)b * N_ + n) * C_ + (size_t)h * 64 + col;
        ao[base + 0]  = (__bf16)(O[u][0][i] * inv);
        ao[base + 16] = (__bf16)(O[u][1][i] * inv);
        ao[base + 32] = (__bf16)(O[u][2][i] * inv);
        ao[base + 48] = (__bf16)(O[u][3][i] * inv);
      }
    }
  }
}

// ---------------------------------------------------------------------------
// Kernel 4: out = attn_out @ Wp + bp (f32 output). One wave = 32x64 tile.
// ---------------------------------------------------------------------------
__global__ __launch_bounds__(32) void outproj_kernel(
    const __bf16* __restrict__ ao, const __bf16* __restrict__ wpt,
    const float* __restrict__ bp, float* __restrict__ out) {
  const int mt = blockIdx.x;              // 0..452
  const int g  = blockIdx.y;              // 0..11 (64-col group)

  v8f acc[2][4];
#pragma unroll
  for (int u = 0; u < 2; ++u)
#pragma unroll
    for (int j = 0; j < 4; ++j) acc[u][j] = (v8f){};

  const __bf16* arow = ao + (size_t)mt * 32 * C_;
  const __bf16* brow = wpt + (size_t)g * 64 * C_;
  for (int kk = 0; kk < C_ / 32; ++kk) {
    v16bf a0 = load_a(arow + kk * 32, C_);
    v16bf a1 = load_a(arow + (size_t)16 * C_ + kk * 32, C_);
#pragma unroll
    for (int j = 0; j < 4; ++j) {
      v16bf bf = load_b(brow + (size_t)j * 16 * C_ + kk * 32, C_);
      acc[0][j] = wmma_bf16(a0, bf, acc[0][j]);
      acc[1][j] = wmma_bf16(a1, bf, acc[1][j]);
    }
  }

  const int lane = threadIdx.x & 31, half = lane >> 4, col = lane & 15;
  float bias[4];
#pragma unroll
  for (int j = 0; j < 4; ++j) bias[j] = bp[g * 64 + j * 16 + col];

#pragma unroll
  for (int u = 0; u < 2; ++u) {
#pragma unroll
    for (int j = 0; j < 4; ++j) {
#pragma unroll
      for (int i = 0; i < 8; ++i) {
        int r = u * 16 + i + 8 * half;
        out[((size_t)mt * 32 + r) * C_ + g * 64 + j * 16 + col] =
            acc[u][j][i] + bias[j];
      }
    }
  }
}

// ---------------------------------------------------------------------------
extern "C" void kernel_launch(void* const* d_in, const int* in_sizes, int n_in,
                              void* d_out, int out_size, void* d_ws, size_t ws_size,
                              hipStream_t stream) {
  const float* x   = (const float*)d_in[0];
  const float* Wq  = (const float*)d_in[1];
  const float* Wk  = (const float*)d_in[2];
  const float* Wv  = (const float*)d_in[3];
  const float* Wvc = (const float*)d_in[4];
  const float* Wp  = (const float*)d_in[5];
  const float* bp  = (const float*)d_in[6];
  float* out = (float*)d_out;

  auto alignup = [](size_t v) { return (v + 255) & ~(size_t)255; };
  char* p = (char*)d_ws;
  __bf16* xb = (__bf16*)p; p += alignup(sizeof(__bf16) * (size_t)B_ * N_ * C_);
  __bf16* wt = (__bf16*)p; p += alignup(sizeof(__bf16) * 5 * (size_t)C_ * C_);
  __bf16* qh = (__bf16*)p; p += alignup(sizeof(__bf16) * (size_t)B_ * H_ * NP_ * D_);
  __bf16* kh = (__bf16*)p; p += alignup(sizeof(__bf16) * (size_t)B_ * H_ * NP_ * D_);
  __bf16* va = (__bf16*)p; p += alignup(sizeof(__bf16) * (size_t)B_ * H_ * D_ * NP_);
  __bf16* vv = (__bf16*)p; p += alignup(sizeof(__bf16) * (size_t)B_ * H_ * D_ * NP_);
  __bf16* ao = (__bf16*)p; p += alignup(sizeof(__bf16) * (size_t)B_ * N_ * C_);
  (void)ws_size; (void)in_sizes; (void)n_in; (void)out_size;

  long ncvt = (long)B_ * N_ * C_ + 5L * C_ * C_;
  int cblocks = (int)((ncvt + 255) / 256);
  cvt_kernel<<<cblocks, 256, 0, stream>>>(x, Wq, Wk, Wv, Wvc, Wp, xb, wt);

  proj_kernel<<<dim3(MT2_, H_, 4), 32, 0, stream>>>(xb, wt, qh, kh, va, vv);

  attn_kernel<<<dim3(30, H_, B_), 32, 0, stream>>>(qh, kh, va, vv, ao);

  outproj_kernel<<<dim3(MT2_, H_), 32, 0, stream>>>(
      ao, wt + 4 * (size_t)C_ * C_, bp, out);
}
